// TorchLSTMRegressor_36069135352383
// MI455X (gfx1250) — matrix operand order repaired and checked by reading, not compiled, not run on hardware
//
#include <hip/hip_runtime.h>
#include <hip/hip_bf16.h>

typedef __attribute__((ext_vector_type(16))) _Float16 v16h;
typedef __attribute__((ext_vector_type(8)))  float    v8f;
typedef __attribute__((ext_vector_type(4)))  unsigned v4u;
typedef __attribute__((ext_vector_type(8)))  int      v8i32;
typedef __attribute__((ext_vector_type(4)))  int      v4i32;

#define SEQ    512
#define BATCH  128
#define NFEAT  64
#define HID    256
#define GATES  1024            // 4*HID
#define NWG    16              // workgroups; each owns HID/NWG = 16 h-columns
#define WGSIZE 256             // 8 waves (wave32)
#define NKREC  8               // 256/32 k-steps for recurrent part
#define NKMAX  16              // layer1: 256/32 + 8
#define WLDS_HALVES (4 * NKMAX * 512)   // 32768 halves = 64 KB LDS weight slab

// ---------------------------------------------------------------------------
// WMMA helpers (CDNA5 gfx1250: wave32, v_wmma_f32_16x16x32_f16)
// ---------------------------------------------------------------------------
__device__ __forceinline__ v8f wmma_f16(v16h a, v16h b, v8f c) {
    return __builtin_amdgcn_wmma_f32_16x16x32_f16(
        /*neg_a=*/false, a, /*neg_b=*/false, b,
        /*c_mod=*/(short)0, c, /*reuse_a=*/false, /*reuse_b=*/false);
}

// A-fragment (16x32 f16, row-major source with given stride in halves).
// lane L -> row L%16; VGPR v half h -> K = (v&3)*2+h + (L>=16)*8 + (v>>2)*16
// -> two contiguous 16B chunks per lane (compiler emits 2x global_load_b128)
__device__ __forceinline__ v16h load_a_frag(const _Float16* base, int stride, int lane) {
    const int row = lane & 15;
    const int hi  = lane >> 4;           // 0 or 1
    const _Float16* p = base + (size_t)row * stride + hi * 8;
    v16h a;
    unsigned* ap = reinterpret_cast<unsigned*>(&a);
#pragma unroll
    for (int v = 0; v < 8; ++v) {
        const int k = ((v & 3) * 2) + ((v >> 2) * 16);
        ap[v] = *reinterpret_cast<const unsigned*>(p + k);
    }
    return a;
}

// B-fragment: packed fragment-major, lane L reads 16 contiguous halves (32B).
__device__ __forceinline__ v16h load_b_frag(const _Float16* frag, int lane) {
    return *reinterpret_cast<const v16h*>(frag + lane * 16);
}

__device__ __forceinline__ float fsigmoid(float x) {
    return 1.0f / (1.0f + __expf(-x));
}
__device__ __forceinline__ float ftanh(float x) {
    return 1.0f - 2.0f / (__expf(2.0f * x) + 1.0f);
}

// ---------------------------------------------------------------------------
// Tensor Data Mover: 1-D slab copy global -> LDS (D# per ISA ch.8)
// Issued by one wave; completion via s_wait_tensorcnt 0.
// 6-arg builtin form (clang-23 / therock-10.0): extra int32x8 group + cpol.
// ---------------------------------------------------------------------------
__device__ __forceinline__ void tdm_load_slab(unsigned lds_off, const void* gptr,
                                              unsigned nelem /* 2-byte elements */) {
    unsigned long long ga = (unsigned long long)(uintptr_t)gptr;
    v4u g0;
    g0[0] = 1u;                                        // count=1, no gather
    g0[1] = lds_off;                                   // lds_addr (bytes)
    g0[2] = (unsigned)(ga & 0xFFFFFFFFu);              // global_addr[31:0]
    g0[3] = (unsigned)((ga >> 32) & 0x01FFFFFFu)       // global_addr[56:32]
          | (2u << 30);                                // type = 2 ("image")
    v8i32 g1;
    g1[0] = (int)(1u << 16);                           // data_size=1 (2B); wg_mask=0
    g1[1] = (int)((nelem & 0xFFFFu) << 16);            // tensor_dim0[15:0]
    g1[2] = (int)(((nelem >> 16) & 0xFFFFu) | (1u << 16)); // tensor_dim0[31:16] | tensor_dim1=1
    g1[3] = (int)((nelem & 0xFFFFu) << 16);            // tile_dim0 (<=65535)
    g1[4] = 1;                                         // tile_dim1 = 1
    g1[5] = (int)nelem;                                // tensor_dim0_stride[31:0]
    g1[6] = 0;
    g1[7] = 0;
    v4i32 z4 = {0, 0, 0, 0};
    v8i32 z8 = {0, 0, 0, 0, 0, 0, 0, 0};
    __builtin_amdgcn_tensor_load_to_lds(g0, g1, z4, z4, z8, 0);
}

// ---------------------------------------------------------------------------
// Grid-wide barrier (sense via monotonically increasing generation counter)
// ---------------------------------------------------------------------------
__device__ __forceinline__ void grid_barrier(unsigned* cnt, unsigned* gen) {
    __threadfence();
    __syncthreads();
    if (threadIdx.x == 0) {
        unsigned g = __hip_atomic_load(gen, __ATOMIC_RELAXED, __HIP_MEMORY_SCOPE_AGENT);
        unsigned arrived = __hip_atomic_fetch_add(cnt, 1u, __ATOMIC_ACQ_REL,
                                                  __HIP_MEMORY_SCOPE_AGENT);
        if (arrived == NWG - 1) {
            __hip_atomic_store(cnt, 0u, __ATOMIC_RELAXED, __HIP_MEMORY_SCOPE_AGENT);
            __hip_atomic_store(gen, g + 1u, __ATOMIC_RELEASE, __HIP_MEMORY_SCOPE_AGENT);
        } else {
            while (__hip_atomic_load(gen, __ATOMIC_ACQUIRE,
                                     __HIP_MEMORY_SCOPE_AGENT) == g) {
                __builtin_amdgcn_s_sleep(2);
            }
        }
    }
    __syncthreads();
}

// ---------------------------------------------------------------------------
// Prep kernels
// ---------------------------------------------------------------------------
__global__ void cvt_x_f16(const float* __restrict__ src, _Float16* __restrict__ dst, int n) {
    for (int i = blockIdx.x * blockDim.x + threadIdx.x; i < n; i += gridDim.x * blockDim.x)
        dst[i] = (_Float16)src[i];
}

// Pack w_ih (4H x K_in) and w_hh (4H x HID) into B-fragment-major f16:
// dst[wg][gate][kstep][512 halves]; lane L -> col n = g*HID + wg*16 + L%16,
// slot s -> k-in-frag = (L>=16)*8 + (s&7) + (s>>3)*16  (ISA B-matrix layout).
__global__ void pack_weights(const float* __restrict__ w_ih, const float* __restrict__ w_hh,
                             int K_in, _Float16* __restrict__ dst, int total) {
    int e = blockIdx.x * blockDim.x + threadIdx.x;
    if (e >= total) return;
    const int kin = K_in / 32;
    const int nk  = kin + NKREC;
    const int r    = e & 511;
    const int frag = e >> 9;
    const int lane = r >> 4;
    const int s    = r & 15;
    const int ks   = frag % nk;
    const int g    = (frag / nk) & 3;
    const int wgi  = frag / (nk * 4);
    const int n    = g * HID + wgi * 16 + (lane & 15);
    const int kk   = ((lane >> 4) << 3) + (s & 7) + ((s >> 3) << 4);
    float v;
    if (ks < kin) v = w_ih[(size_t)n * K_in + ks * 32 + kk];
    else          v = w_hh[(size_t)n * HID + (ks - kin) * 32 + kk];
    dst[e] = (_Float16)v;
}

__global__ void combine_bias(const float* __restrict__ b_ih, const float* __restrict__ b_hh,
                             float* __restrict__ dst, int n) {
    int i = blockIdx.x * blockDim.x + threadIdx.x;
    if (i < n) dst[i] = b_ih[i] + b_hh[i];
}

__global__ void init_barrier(unsigned* bar) {
    if (threadIdx.x < 2) bar[threadIdx.x] = 0u;
}

// ---------------------------------------------------------------------------
// One LSTM layer, persistent across SEQ steps with per-step grid barrier.
// B-fragments served from the LDS slab staged by the TDM.
// ---------------------------------------------------------------------------
__device__ __forceinline__ void run_layer(
    const _Float16* __restrict__ in_buf, int C_in, int nkIn,
    const _Float16* __restrict__ pW, const float* __restrict__ bias,
    _Float16* __restrict__ out_buf,
    _Float16* w_lds, float* c_lds, int wg, int wave, int lane,
    unsigned* bar_cnt, unsigned* bar_gen)
{
    const int nk  = nkIn + NKREC;
    const int col = lane & 15;
    const int hi  = lane >> 4;

    // Stage this WG's weight slab (4*nk*512 halves) into LDS via the TDM.
    if (threadIdx.x < 32) {
        tdm_load_slab((unsigned)(uintptr_t)w_lds,
                      pW + (size_t)wg * 4 * nk * 512,
                      (unsigned)(4 * nk * 512));
        __builtin_amdgcn_s_wait_tensorcnt(0);
    }

    // Per-lane bias (fixed across steps/rows)
    const float bi = bias[0 * HID + wg * 16 + col];
    const float bf = bias[1 * HID + wg * 16 + col];
    const float bg = bias[2 * HID + wg * 16 + col];
    const float bo = bias[3 * HID + wg * 16 + col];

    // Per-gate LDS fragment bases
    const _Float16* wl0 = w_lds + (size_t)0 * nk * 512;
    const _Float16* wl1 = w_lds + (size_t)1 * nk * 512;
    const _Float16* wl2 = w_lds + (size_t)2 * nk * 512;
    const _Float16* wl3 = w_lds + (size_t)3 * nk * 512;

    // c = 0
    for (int i = threadIdx.x; i < BATCH * 16; i += WGSIZE) c_lds[i] = 0.0f;
    __syncthreads();   // publishes TDM slab + zeroed c to all waves

    for (int t = 0; t < SEQ; ++t) {
        v8f a0 = {}, a1 = {}, a2 = {}, a3 = {};

        // ---- input projection: x_t @ w_ih^T ----
        const _Float16* abase = in_buf + ((size_t)t * BATCH + wave * 16) * C_in;
        for (int ks = 0; ks < nkIn; ++ks) {
            v16h a = load_a_frag(abase + ks * 32, C_in, lane);
            a0 = wmma_f16(a, load_b_frag(wl0 + (size_t)ks * 512, lane), a0);
            a1 = wmma_f16(a, load_b_frag(wl1 + (size_t)ks * 512, lane), a1);
            a2 = wmma_f16(a, load_b_frag(wl2 + (size_t)ks * 512, lane), a2);
            a3 = wmma_f16(a, load_b_frag(wl3 + (size_t)ks * 512, lane), a3);
        }
        // ---- recurrence: h_{t-1} @ w_hh^T (h_0 = 0 -> skip at t==0) ----
        if (t > 0) {
            const _Float16* hbase = out_buf + ((size_t)(t - 1) * BATCH + wave * 16) * HID;
#pragma unroll
            for (int ks = 0; ks < NKREC; ++ks) {
                v16h a = load_a_frag(hbase + ks * 32, HID, lane);
                a0 = wmma_f16(a, load_b_frag(wl0 + (size_t)(nkIn + ks) * 512, lane), a0);
                a1 = wmma_f16(a, load_b_frag(wl1 + (size_t)(nkIn + ks) * 512, lane), a1);
                a2 = wmma_f16(a, load_b_frag(wl2 + (size_t)(nkIn + ks) * 512, lane), a2);
                a3 = wmma_f16(a, load_b_frag(wl3 + (size_t)(nkIn + ks) * 512, lane), a3);
            }
        }
        // Prefetch next timestep's activation rows (global_prefetch_b8)
        if (t + 1 < SEQ) {
            const _Float16* pf = in_buf + ((size_t)(t + 1) * BATCH + wave * 16 + col) * C_in;
            __builtin_prefetch(pf, 0, 1);
        }

        // ---- LSTM cell update (gate order i, f, g, o) ----
        const int rowbase = wave * 16 + hi * 8;
        _Float16* hout = out_buf + (size_t)t * BATCH * HID + (size_t)wg * 16 + col;
#pragma unroll
        for (int v = 0; v < 8; ++v) {
            const int r  = rowbase + v;
            const int ci = r * 16 + col;
            float gi = fsigmoid(a0[v] + bi);
            float gf = fsigmoid(a1[v] + bf);
            float gg = ftanh   (a2[v] + bg);
            float go = fsigmoid(a3[v] + bo);
            float c  = gf * c_lds[ci] + gi * gg;
            c_lds[ci] = c;
            float h  = go * ftanh(c);
            hout[(size_t)r * HID] = (_Float16)h;
        }
        grid_barrier(bar_cnt, bar_gen);   // publish h_t before anyone reads it
    }
}

// ---------------------------------------------------------------------------
// Main persistent kernel: layer0 -> layer1 -> FC head
// ---------------------------------------------------------------------------
__global__ __launch_bounds__(WGSIZE)
void lstm2_persistent(const _Float16* __restrict__ X16,
                      const _Float16* __restrict__ pW0,
                      const _Float16* __restrict__ pW1,
                      const float*    __restrict__ bias0,
                      const float*    __restrict__ bias1,
                      _Float16*       __restrict__ out0,
                      _Float16*       __restrict__ out1,
                      const float*    __restrict__ fc_w,
                      const float*    __restrict__ fc_b,
                      float*          __restrict__ dout,
                      unsigned*       __restrict__ bar)
{
    __shared__ __align__(128) _Float16 w_lds[WLDS_HALVES];  // 64 KB weight slab
    __shared__ float c_lds[BATCH * 16];                     // 8 KB cell-state slice
    const int wg   = blockIdx.x;
    const int lane = threadIdx.x & 31;
    const int wave = threadIdx.x >> 5;

    run_layer(X16,  NFEAT, NFEAT / 32, pW0, bias0, out0,
              w_lds, c_lds, wg, wave, lane, bar, bar + 1);
    run_layer(out0, HID,   HID / 32,   pW1, bias1, out1,
              w_lds, c_lds, wg, wave, lane, bar, bar + 1);

    // FC head: d_out[r] = out1[last] . fc_w + fc_b
    if (wg == 0 && threadIdx.x < BATCH) {
        const int r = threadIdx.x;
        const _Float16* h = out1 + ((size_t)(SEQ - 1) * BATCH + r) * HID;
        float s = 0.0f;
#pragma unroll 8
        for (int k = 0; k < HID; ++k) s += (float)h[k] * fc_w[k];
        dout[r] = s + fc_b[0];
    }
}

// ---------------------------------------------------------------------------
// Host launch
// ---------------------------------------------------------------------------
extern "C" void kernel_launch(void* const* d_in, const int* in_sizes, int n_in,
                              void* d_out, int out_size, void* d_ws, size_t ws_size,
                              hipStream_t stream) {
    (void)in_sizes; (void)n_in; (void)out_size; (void)ws_size;
    const float* X     = (const float*)d_in[0];
    const float* w_ih0 = (const float*)d_in[1];
    const float* w_hh0 = (const float*)d_in[2];
    const float* b_ih0 = (const float*)d_in[3];
    const float* b_hh0 = (const float*)d_in[4];
    const float* w_ih1 = (const float*)d_in[5];
    const float* w_hh1 = (const float*)d_in[6];
    const float* b_ih1 = (const float*)d_in[7];
    const float* b_hh1 = (const float*)d_in[8];
    const float* fc_w  = (const float*)d_in[9];
    const float* fc_b  = (const float*)d_in[10];
    float* out = (float*)d_out;

    char*  ws  = (char*)d_ws;
    size_t off = 0;
    auto take = [&](size_t bytes) -> void* {
        void* p = ws + off;
        off += (bytes + 255) & ~(size_t)255;
        return p;
    };
    unsigned*  bar   = (unsigned*)take(256);
    _Float16*  X16   = (_Float16*)take((size_t)SEQ * BATCH * NFEAT * sizeof(_Float16));
    _Float16*  out0  = (_Float16*)take((size_t)SEQ * BATCH * HID * sizeof(_Float16));
    _Float16*  out1  = (_Float16*)take((size_t)SEQ * BATCH * HID * sizeof(_Float16));
    const int  nk0   = NFEAT / 32 + NKREC;                 // 10
    const int  nk1   = HID / 32 + NKREC;                   // 16
    const int  tot0  = NWG * 4 * nk0 * 512;
    const int  tot1  = NWG * 4 * nk1 * 512;
    _Float16*  pW0   = (_Float16*)take((size_t)tot0 * sizeof(_Float16));
    _Float16*  pW1   = (_Float16*)take((size_t)tot1 * sizeof(_Float16));
    float*     bias0 = (float*)take(GATES * sizeof(float));
    float*     bias1 = (float*)take(GATES * sizeof(float));

    init_barrier<<<1, 32, 0, stream>>>(bar);
    {
        const int n = SEQ * BATCH * NFEAT;
        cvt_x_f16<<<(n + 1023) / 1024, 1024, 0, stream>>>(X, X16, n);
    }
    pack_weights<<<(tot0 + 255) / 256, 256, 0, stream>>>(w_ih0, w_hh0, NFEAT, pW0, tot0);
    pack_weights<<<(tot1 + 255) / 256, 256, 0, stream>>>(w_ih1, w_hh1, HID,   pW1, tot1);
    combine_bias<<<(GATES + 255) / 256, 256, 0, stream>>>(b_ih0, b_hh0, bias0, GATES);
    combine_bias<<<(GATES + 255) / 256, 256, 0, stream>>>(b_ih1, b_hh1, bias1, GATES);

    lstm2_persistent<<<NWG, WGSIZE, 0, stream>>>(X16, pW0, pW1, bias0, bias1,
                                                 out0, out1, fc_w, fc_b, out, bar);
}
